// CrossAttention_88407606820965
// MI455X (gfx1250) — compile-verified, matrix operand order checked by reading, not run on hardware
//
#include <hip/hip_runtime.h>
#include <hip/hip_bf16.h>

typedef __attribute__((ext_vector_type(16))) _Float16 v16h;
typedef __attribute__((ext_vector_type(8)))  _Float16 v8h;
typedef __attribute__((ext_vector_type(2)))  _Float16 v2h;
typedef __attribute__((ext_vector_type(8)))  float    v8f;

#define HW   16384   // 128*128
#define WDIM 128
#define CDIM 384

union V16u { v16h v; struct { v8h lo; v8h hi; } p; };

// Load a 16-half WMMA A/B fragment: lane supplies base = row/col ptr + kbase.
// Halves 0..7 from [base], halves 8..15 from [base+16] (K-groups per ISA layout).
__device__ __forceinline__ v16h load_frag(const _Float16* base) {
    V16u u;
    const v8h* p = (const v8h*)base;
    u.p.lo = p[0];
    u.p.hi = p[2];   // +16 halfs
    return u.v;
}

__device__ __forceinline__ v8f wmma_f16(v16h a, v16h b, v8f c) {
    return __builtin_amdgcn_wmma_f32_16x16x32_f16(false, a, false, b, (short)0, c,
                                                  false, false);
}

__global__ void f32_to_f16_kernel(const float* __restrict__ a,
                                  _Float16* __restrict__ o, int n) {
    int i = blockIdx.x * 256 + threadIdx.x;
    if (i < n) o[i] = (_Float16)a[i];
}

// Channel LayerNorm of one 8x8 window into LDS as fp16, layout [xy][c].
// Reads global exactly once: raw f32 values are staged into s_raw (LDS) during
// the stats pass and normalized from LDS. Ends with a barrier (s_raw recycled).
__device__ void window_ln(const float* __restrict__ src, size_t gbase,
                          const float* __restrict__ gamma, const float* __restrict__ beta,
                          _Float16* dst, float* s_raw,
                          float (*s_red)[6][64], float* s_mean, float* s_rstd,
                          int tid) {
    {   // partial stats: 6 threads per xy column, 64 channels each; stage raw to LDS
        int xy = tid & 63, part = tid >> 6;
        int dy = xy >> 3, dx = xy & 7;
        const float* px = src + gbase + (size_t)(part * 64) * HW + dy * WDIM + dx;
        float* rw = s_raw + (part * 64) * 64 + xy;    // raw layout: [c][xy]
        float s = 0.f, sq = 0.f;
        #pragma unroll 4
        for (int c = 0; c < 64; ++c) {
            float v = px[(size_t)c * HW];
            rw[c * 64] = v;
            s += v; sq += v * v;
        }
        s_red[0][part][xy] = s;
        s_red[1][part][xy] = sq;
    }
    __syncthreads();
    if (tid < 64) {
        float s = 0.f, sq = 0.f;
        for (int pt = 0; pt < 6; ++pt) { s += s_red[0][pt][tid]; sq += s_red[1][pt][tid]; }
        float mean = s * (1.f / 384.f);
        float var  = sq * (1.f / 384.f) - mean * mean;
        s_mean[tid] = mean;
        s_rstd[tid] = rsqrtf(var + 1e-5f);
    }
    __syncthreads();
    for (int i = tid; i < 384 * 64; i += 384) {     // i = c*64 + p  == raw index
        int c = i >> 6, p = i & 63;
        float v = s_raw[i];
        dst[p * 384 + c] =
            (_Float16)((v - s_mean[p]) * s_rstd[p] * gamma[c] + beta[c]);
    }
    __syncthreads();   // s_raw / s_mean recycled by the caller next
}

__global__ __launch_bounds__(384, 1)
void win_xattn_kernel(const float* __restrict__ x, const float* __restrict__ skip,
                      const float* __restrict__ g_q,  const float* __restrict__ b_q,
                      const float* __restrict__ g_kv, const float* __restrict__ b_kv,
                      const _Float16* __restrict__ Wq_h,
                      const _Float16* __restrict__ Wkv_h,
                      const _Float16* __restrict__ Wout_h,
                      const float* __restrict__ bout,
                      float* __restrict__ out) {
    // [xy][c] fp16 operand tiles; s_xn is reused as attention output [xy][inner]
    __shared__ _Float16 s_xn [64 * 384];
    __shared__ _Float16 s_skn[64 * 384];
    // s_q(48K) + s_k(48K); same bytes host the raw f32 window (96K) during LN
    __shared__ __align__(16) unsigned char s_qk_bytes[98304];
    __shared__ _Float16 s_v  [384 * 64];   // [o][xy]  (xy contiguous)
    __shared__ float s_red[2][6][64];
    __shared__ float s_mean[64];
    __shared__ float s_rstd[64];

    _Float16* s_q  = (_Float16*)s_qk_bytes;            // [xy][o] (o contiguous)
    _Float16* s_k  = (_Float16*)(s_qk_bytes + 49152);  // [xy][o]
    float*    s_raw = (float*)s_qk_bytes;              // LN staging, [c][xy]

    const int tid  = threadIdx.x;
    const int lane = tid & 31;
    const int wv   = tid >> 5;        // wave 0..11
    const int lrow = lane & 15;       // row (A) / col (B,C,D) within 16x16 tile
    const int lh   = lane >> 4;       // lane-half
    const int kbase = lh * 8;         // K-group base for 16-bit A/B fragments

    const int win = blockIdx.x;       // 2048 windows
    const int bi  = win >> 8;
    const int wy  = (win >> 4) & 15;
    const int wx  = win & 15;
    const int y0  = wy * 8, x0 = wx * 8;
    const size_t gbase = (size_t)bi * CDIM * HW + (size_t)y0 * WDIM + x0;

    // Warm the Wq stream (first GEMM) while LN runs: global_prefetch_b8
    for (int i = tid; i < 2304; i += 384)                 // 288KB / 128B lines
        __builtin_prefetch((const char*)Wq_h + (size_t)i * 128, 0, 3);

    // ---- Phase 0: channel LN of x and skip into LDS (fp16), single HBM read each ----
    window_ln(x,    gbase, g_q,  b_q,  s_xn,  s_raw, s_red, s_mean, s_rstd, tid);
    window_ln(skip, gbase, g_kv, b_kv, s_skn, s_raw, s_red, s_mean, s_rstd, tid);

    // ---- Phase 1: q = Wq * xn   (24 o-tiles x 4 xy-tiles) ----
    for (int t = wv; t < 96; t += 12) {
        int ot = t >> 2, nt = t & 3;
        v8f acc = {};
        const _Float16* arow = Wq_h + (size_t)(ot * 16 + lrow) * 384 + kbase;
        const _Float16* brow = s_xn + (nt * 16 + lrow) * 384 + kbase;
        #pragma unroll
        for (int kk = 0; kk < 12; ++kk)
            acc = wmma_f16(load_frag(arow + kk * 32), load_frag(brow + kk * 32), acc);
        int p  = nt * 16 + lrow;
        int ob = ot * 16 + lh * 8;
        #pragma unroll
        for (int j = 0; j < 8; j += 2) {
            v2h pr; pr.x = (_Float16)acc[j]; pr.y = (_Float16)acc[j + 1];
            *(v2h*)&s_q[p * 384 + ob + j] = pr;
        }
    }

    // ---- Phase 2: kv = Wkv * skn  (48 o-tiles x 4 xy-tiles) ----
    for (int t = wv; t < 192; t += 12) {
        int ot = t >> 2, nt = t & 3;
        v8f acc = {};
        const _Float16* arow = Wkv_h + (size_t)(ot * 16 + lrow) * 384 + kbase;
        const _Float16* brow = s_skn + (nt * 16 + lrow) * 384 + kbase;
        #pragma unroll
        for (int kk = 0; kk < 12; ++kk)
            acc = wmma_f16(load_frag(arow + kk * 32), load_frag(brow + kk * 32), acc);
        int p  = nt * 16 + lrow;
        int ob = ot * 16 + lh * 8;
        if (ob < 384) {           // K half -> [xy][o]
            #pragma unroll
            for (int j = 0; j < 8; j += 2) {
                v2h pr; pr.x = (_Float16)acc[j]; pr.y = (_Float16)acc[j + 1];
                *(v2h*)&s_k[p * 384 + ob + j] = pr;
            }
        } else {                  // V half -> [o][xy]
            #pragma unroll
            for (int j = 0; j < 8; ++j)
                s_v[(ob - 384 + j) * 64 + p] = (_Float16)acc[j];
        }
    }
    __syncthreads();

    // ---- Phase 3: attention, one head per wave (h = wv, DH=32 => K=32 single WMMA) ----
    {
        const int hb = wv * 32;
        v16h ak[4], bq[4];
        #pragma unroll
        for (int mt = 0; mt < 4; ++mt)     // A = k^T : row m=k_xy, K=d contiguous
            ak[mt] = load_frag(s_k + (mt * 16 + lrow) * 384 + hb + kbase);
        #pragma unroll
        for (int nt = 0; nt < 4; ++nt)     // B = q   : col n=q_xy, K=d contiguous
            bq[nt] = load_frag(s_q + (nt * 16 + lrow) * 384 + hb + kbase);

        v8f S[4][4];                        // S^T tiles [k_xy-tile][q_xy-tile]
        #pragma unroll
        for (int mt = 0; mt < 4; ++mt)
            #pragma unroll
            for (int nt = 0; nt < 4; ++nt) {
                v8f z = {};
                S[mt][nt] = wmma_f16(ak[mt], bq[nt], z);
            }

        // softmax over k (rows of S^T) per q column; lane holds 32 rows, lane^16 the rest
        const float scale = 0.17677669529663687f;   // 1/sqrt(32)
        #pragma unroll
        for (int nt = 0; nt < 4; ++nt) {
            float mx = -3.0e38f;
            #pragma unroll
            for (int mt = 0; mt < 4; ++mt)
                #pragma unroll
                for (int j = 0; j < 8; ++j) mx = fmaxf(mx, S[mt][nt][j]);
            mx = fmaxf(mx, __shfl_xor(mx, 16, 32));
            float sum = 0.f;
            #pragma unroll
            for (int mt = 0; mt < 4; ++mt)
                #pragma unroll
                for (int j = 0; j < 8; ++j) {
                    float e = __expf((S[mt][nt][j] - mx) * scale);
                    S[mt][nt][j] = e;
                    sum += e;
                }
            sum += __shfl_xor(sum, 16, 32);
            float r = 1.0f / sum;
            #pragma unroll
            for (int mt = 0; mt < 4; ++mt)
                #pragma unroll
                for (int j = 0; j < 8; ++j) S[mt][nt][j] *= r;
        }

        // P^T accumulators -> B fragments, lane-locally (rows line up with K-groups)
        v16h bp[2][4];
        #pragma unroll
        for (int kk = 0; kk < 2; ++kk)
            #pragma unroll
            for (int nt = 0; nt < 4; ++nt) {
                v16h tf;
                #pragma unroll
                for (int q = 0; q < 8; ++q) {
                    tf[q]     = (_Float16)S[2 * kk][nt][q];
                    tf[q + 8] = (_Float16)S[2 * kk + 1][nt][q];
                }
                bp[kk][nt] = tf;
            }

        // A = v : row m=d, K=k_xy contiguous
        v16h av[2][2];
        #pragma unroll
        for (int mt = 0; mt < 2; ++mt)
            #pragma unroll
            for (int kk = 0; kk < 2; ++kk)
                av[mt][kk] = load_frag(s_v + (hb + mt * 16 + lrow) * 64 + kk * 32 + kbase);

        // out[d, q] = v @ P^T, write to s_att (= s_xn) as [q_xy][inner]
        #pragma unroll
        for (int mt = 0; mt < 2; ++mt)
            #pragma unroll
            for (int nt = 0; nt < 4; ++nt) {
                v8f o = {};
                o = wmma_f16(av[mt][0], bp[0][nt], o);
                o = wmma_f16(av[mt][1], bp[1][nt], o);
                int p  = nt * 16 + lrow;
                int db = hb + mt * 16 + lh * 8;
                #pragma unroll
                for (int j = 0; j < 8; j += 2) {
                    v2h pr; pr.x = (_Float16)o[j]; pr.y = (_Float16)o[j + 1];
                    *(v2h*)&s_xn[p * 384 + db + j] = pr;
                }
            }
    }
    __syncthreads();

    // ---- Phase 4: y = Wout * att + bout, scatter back to [b][c][y][x] ----
    for (int t = wv; t < 96; t += 12) {
        int ot = t >> 2, nt = t & 3;
        v8f acc = {};
        const _Float16* arow = Wout_h + (size_t)(ot * 16 + lrow) * 384 + kbase;
        const _Float16* brow = s_xn + (nt * 16 + lrow) * 384 + kbase;
        #pragma unroll
        for (int kk = 0; kk < 12; ++kk)
            acc = wmma_f16(load_frag(arow + kk * 32), load_frag(brow + kk * 32), acc);
        int p  = nt * 16 + lrow;
        int dy = p >> 3, dx = p & 7;
        #pragma unroll
        for (int j = 0; j < 8; ++j) {
            int o = ot * 16 + lh * 8 + j;
            out[gbase + (size_t)o * HW + dy * WDIM + dx] = acc[j] + bout[o];
        }
    }
}

extern "C" void kernel_launch(void* const* d_in, const int* in_sizes, int n_in,
                              void* d_out, int out_size, void* d_ws, size_t ws_size,
                              hipStream_t stream) {
    const float* x    = (const float*)d_in[0];
    const float* skip = (const float*)d_in[1];
    const float* g_q  = (const float*)d_in[2];
    const float* b_q  = (const float*)d_in[3];
    const float* g_kv = (const float*)d_in[4];
    const float* b_kv = (const float*)d_in[5];
    const float* Wq   = (const float*)d_in[6];
    const float* Wkv  = (const float*)d_in[7];
    const float* Wout = (const float*)d_in[8];
    const float* bout = (const float*)d_in[9];
    float* out = (float*)d_out;

    _Float16* wqh   = (_Float16*)d_ws;
    _Float16* wkvh  = wqh + 384 * 384;
    _Float16* wouth = wkvh + 768 * 384;

    f32_to_f16_kernel<<<(384 * 384 + 255) / 256, 256, 0, stream>>>(Wq, wqh, 384 * 384);
    f32_to_f16_kernel<<<(768 * 384 + 255) / 256, 256, 0, stream>>>(Wkv, wkvh, 768 * 384);
    f32_to_f16_kernel<<<(384 * 384 + 255) / 256, 256, 0, stream>>>(Wout, wouth, 384 * 384);

    win_xattn_kernel<<<2048, 384, 0, stream>>>(x, skip, g_q, b_q, g_kv, b_kv,
                                               wqh, wkvh, wouth, bout, out);
}
